// DeltaSepConvGRU_41248865911330
// MI455X (gfx1250) — matrix operand order repaired and checked by reading, not compiled, not run on hardware
//
#include <hip/hip_runtime.h>
#include <math.h>

// ---------------- problem constants ----------------
#define NB    4
#define HID   128
#define INP   320
#define CINN  448          // HID + INP
#define HH    64
#define WW    160
#define HWSZ  (HH*WW)      // 10240
#define PIXT  (NB*HWSZ)    // 40960 total output pixels
#define TAPS  5
#define CHUNKS 14          // 448 / 32
#define NT    32           // pixel tile width
#define SSTR  40           // slab row stride in elements (80B: 16B-aligned, bank-spread)
#define THRESHV 0.02f

typedef __bf16 bf16_t;
typedef __attribute__((ext_vector_type(16))) __bf16 v16bf;
typedef __attribute__((ext_vector_type(8)))  float  v8f;

union Frag16 { v16bf v; unsigned u[8]; uint4 q[2]; };

__device__ __forceinline__ float sigf(float v) { return 1.f / (1.f + expf(-v)); }

// ============================================================================
// Stage one 32-channel input slab into LDS, transposed to [pixelIdx][ci],
// dword-vectorized (bf16 pairs share in/out-of-bounds status: tile origins and
// image bounds are even). Zero-fills the conv's SAME padding.
// ============================================================================
template<bool VERT>
__device__ __forceinline__ void stage_slab(const bf16_t* __restrict__ in, bf16_t* __restrict__ sl,
                                           int tid, int nb, int row, int col0, int c0)
{
    if (VERT) {
        // 5 tap-rows x 32 ch x 16 dword pairs
        for (int i = tid; i < TAPS * 32 * 16; i += 256) {
            const int pr = i & 15;
            const int ci = (i >> 4) & 31;
            const int t  = i >> 9;
            const int r  = row + t - 2;
            unsigned w = 0u;
            if (r >= 0 && r < HH)
                w = *(const unsigned*)&in[(((size_t)nb * CINN + (c0 + ci)) * HH + r) * WW + col0 + 2 * pr];
            const int px = t * NT + 2 * pr;
            *(unsigned short*)&sl[px * SSTR + ci]       = (unsigned short)(w & 0xffffu);
            *(unsigned short*)&sl[(px + 1) * SSTR + ci] = (unsigned short)(w >> 16);
        }
    } else {
        // 32 ch x 18 dword pairs (36 px incl. +-2 halo)
        for (int i = tid; i < 32 * 18; i += 256) {
            const int pr  = i % 18;
            const int ci  = i / 18;
            const int col = col0 - 2 + 2 * pr;
            unsigned w = 0u;
            if (col >= 0 && col < WW)
                w = *(const unsigned*)&in[(((size_t)nb * CINN + (c0 + ci)) * HH + row) * WW + col];
            const int px = 2 * pr;
            *(unsigned short*)&sl[px * SSTR + ci]       = (unsigned short)(w & 0xffffu);
            *(unsigned short*)&sl[(px + 1) * SSTR + ci] = (unsigned short)(w >> 16);
        }
    }
}

// ============================================================================
// Implicit-GEMM separable conv via WMMA bf16 (f32 accumulate).
//   out[oc, n,row,col] = sum_t sum_ci W[oc][ci][t] * in[ci, shifted(t)]
// VERT=0 : 1x5 conv (taps shift along W, pad 2); VERT=1 : 5x1 (taps along H).
// Tile: 128 oc x 32 px per workgroup (8 waves); each wave: 16 oc rows x both
// 16-px N-subtiles. K-loop: 14 chunks of 32 ci x 5 taps, double-buffered slab.
// A fragments: 2x b128 from pre-packed weights; B fragments: 2x ds b128
// (the B layout K = half*16 + 2v makes each fragment a contiguous 32B run).
// ============================================================================
template<bool VERT>
__global__ __launch_bounds__(256)
void conv_wmma(const bf16_t* __restrict__ in, const bf16_t* __restrict__ wp,
               const float* __restrict__ bias, const float* __restrict__ accum,
               float* __restrict__ out, int OC)
{
    constexpr int SLAB_PX = VERT ? (TAPS * NT) : (NT + 4);
    __shared__ bf16_t slab[2][SLAB_PX * SSTR];

    const int tid  = threadIdx.x;
    const int wid  = tid >> 5;
    const int lane = tid & 31;
    const int half = lane >> 4;   // which 16-lane half of the wave
    const int nl   = lane & 15;

    const int cb   = WW / NT;     // 5 column blocks per row
    const int pb   = blockIdx.x;
    const int col0 = (pb % cb) * NT;
    const int row  = (pb / cb) % HH;
    const int nb   =  pb / (cb * HH);
    const int ocBlock = blockIdx.y * 128;
    const int ocSub   = ocBlock + wid * 16 + nl;   // A-matrix row for this lane

    v8f acc0 = {}; v8f acc1 = {};

    stage_slab<VERT>(in, slab[0], tid, nb, row, col0, 0);
    __syncthreads();

    for (int c = 0; c < CHUNKS; ++c) {
        if (c + 1 < CHUNKS)
            stage_slab<VERT>(in, slab[(c + 1) & 1], tid, nb, row, col0, (c + 1) * 32);

        const bf16_t* sl = slab[c & 1];
        #pragma unroll
        for (int t = 0; t < TAPS; ++t) {
            // A fragment: 32 contiguous bytes per lane from the packed weights.
            Frag16 a;
            const uint4* ap = (const uint4*)(wp +
                ((((size_t)t * CHUNKS + c) * OC + ocSub) << 5) + (half << 4));
            a.q[0] = ap[0]; a.q[1] = ap[1];

            // B fragments: contiguous 32B per lane from the LDS slab.
            Frag16 b0, b1;
            const int p0 = VERT ? (t * NT + nl)      : (nl + t);
            const int p1 = VERT ? (t * NT + 16 + nl) : (16 + nl + t);
            const uint4* bp0 = (const uint4*)(sl + p0 * SSTR + (half << 4));
            const uint4* bp1 = (const uint4*)(sl + p1 * SSTR + (half << 4));
            b0.q[0] = bp0[0]; b0.q[1] = bp0[1];
            b1.q[0] = bp1[0]; b1.q[1] = bp1[1];

            acc0 = __builtin_amdgcn_wmma_f32_16x16x32_bf16(false, a.v, false, b0.v,
                                                           (short)0, acc0, false, false);
            acc1 = __builtin_amdgcn_wmma_f32_16x16x32_bf16(false, a.v, false, b1.v,
                                                           (short)0, acc1, false, false);
        }
        __syncthreads();
    }

    // ---- writeout: D row M = half*8 + r, column N = nl ----
    const int ocOut = ocBlock + wid * 16 + half * 8;
    #pragma unroll
    for (int r = 0; r < 8; ++r) {
        const int oc = ocOut + r;
        const float bv = bias ? bias[oc] : 0.f;
        const size_t base = (((size_t)nb * OC + oc) * HH + row) * WW + col0;
        const size_t i0 = base + nl;
        const size_t i1 = base + 16 + nl;
        float v0 = acc0[r] + bv;
        float v1 = acc1[r] + bv;
        if (accum) { v0 += accum[i0]; v1 += accum[i1]; }
        out[i0] = v0;
        out[i1] = v1;
    }
}

// ============================================================================
// Weight re-pack: f32 W[oc][448][5] -> bf16, A-fragment-ordered:
//   wp[((t*14+c)*OCtot + ocOff+oc)*32 + j]; per-lane half holds
//   K = half*16 + v*2 + off, with j = half*16 + v*2 + off
// ============================================================================
__global__ void pack_weights(const float* __restrict__ w, bf16_t* __restrict__ wp,
                             int ocOff, int OCtot)
{
    const int TOT = TAPS * CHUNKS * HID * 32;
    int i = blockIdx.x * 256 + threadIdx.x;
    if (i >= TOT) return;
    const int j   = i & 31;
    const int oc  = (i >> 5) & 127;
    const int c   = (i >> 12) % CHUNKS;
    const int t   = i / (CHUNKS << 12);
    const int hlf = j >> 4;
    const int rr  = j & 15;
    const int v   = rr >> 1;
    const int off = rr & 1;
    const int K   = (hlf << 4) | rr;
    const int ci  = c * 32 + K;
    const float val = w[((size_t)oc * CINN + ci) * TAPS + t];
    wp[((((size_t)t * CHUNKS + c) * OCtot + ocOff + oc) << 5) + j] = (bf16_t)val;
}

__global__ void stack_bias(const float* __restrict__ b0, const float* __restrict__ b1,
                           float* __restrict__ dst)
{
    int i = threadIdx.x;
    if (i < HID) dst[i] = b0[i];
    else         dst[i] = b1[i - HID];
}

// ============================================================================
// Elementwise kernels (all f32 math; bf16 only for conv operands / mem concat)
// ============================================================================
__global__ void concat_init(const float* __restrict__ h, const float* __restrict__ x,
                            bf16_t* __restrict__ mem)
{
    size_t i = (size_t)blockIdx.x * 256 + threadIdx.x;
    if (i >= (size_t)NB * CINN * HWSZ) return;
    const int p  = (int)(i % HWSZ);
    const int ch = (int)((i / HWSZ) % CINN);
    const int n  = (int)(i / ((size_t)CINN * HWSZ));
    const float v = (ch < HID) ? h[((size_t)n * HID + ch) * HWSZ + p]
                               : x[((size_t)n * INP + (ch - HID)) * HWSZ + p];
    mem[i] = (bf16_t)v;
}

__global__ void concat_delta(const float* __restrict__ h, const float* __restrict__ x,
                             const bf16_t* __restrict__ mem, bf16_t* __restrict__ dout)
{
    size_t i = (size_t)blockIdx.x * 256 + threadIdx.x;
    if (i >= (size_t)NB * CINN * HWSZ) return;
    const int p  = (int)(i % HWSZ);
    const int ch = (int)((i / HWSZ) % CINN);
    const int n  = (int)(i / ((size_t)CINN * HWSZ));
    const float v = (ch < HID) ? h[((size_t)n * HID + ch) * HWSZ + p]
                               : x[((size_t)n * INP + (ch - HID)) * HWSZ + p];
    const float d = v - (float)mem[i];
    dout[i] = (fabsf(d) <= THRESHV) ? (bf16_t)0.f : (bf16_t)d;
}

// rhx = concat(sigmoid(r)*h, x); czr holds [z(0..127) | r(128..255)] channels.
__global__ void rhx_init(const float* __restrict__ czr, const float* __restrict__ h,
                         const float* __restrict__ x, bf16_t* __restrict__ mem)
{
    size_t i = (size_t)blockIdx.x * 256 + threadIdx.x;
    if (i >= (size_t)NB * CINN * HWSZ) return;
    const int p  = (int)(i % HWSZ);
    const int ch = (int)((i / HWSZ) % CINN);
    const int n  = (int)(i / ((size_t)CINN * HWSZ));
    float v;
    if (ch < HID) {
        const float r = sigf(czr[((size_t)n * 256 + HID + ch) * HWSZ + p]);
        v = r * h[((size_t)n * HID + ch) * HWSZ + p];
    } else {
        v = x[((size_t)n * INP + (ch - HID)) * HWSZ + p];
    }
    mem[i] = (bf16_t)v;
}

__global__ void rhx_delta(const float* __restrict__ czr, const float* __restrict__ h,
                          const float* __restrict__ x, const bf16_t* __restrict__ mem,
                          bf16_t* __restrict__ dout)
{
    size_t i = (size_t)blockIdx.x * 256 + threadIdx.x;
    if (i >= (size_t)NB * CINN * HWSZ) return;
    const int p  = (int)(i % HWSZ);
    const int ch = (int)((i / HWSZ) % CINN);
    const int n  = (int)(i / ((size_t)CINN * HWSZ));
    float v;
    if (ch < HID) {
        const float r = sigf(czr[((size_t)n * 256 + HID + ch) * HWSZ + p]);
        v = r * h[((size_t)n * HID + ch) * HWSZ + p];
    } else {
        v = x[((size_t)n * INP + (ch - HID)) * HWSZ + p];
    }
    const float d = v - (float)mem[i];
    dout[i] = (fabsf(d) <= THRESHV) ? (bf16_t)0.f : (bf16_t)d;
}

// h_out = (1-z)*h_in + z*tanh(cq),  z = sigmoid(czr[z-part])
__global__ void gru_update(const float* __restrict__ czr, const float* __restrict__ cq,
                           const float* __restrict__ hin, float* __restrict__ hout)
{
    size_t i = (size_t)blockIdx.x * 256 + threadIdx.x;
    if (i >= (size_t)NB * HID * HWSZ) return;
    const int p  = (int)(i % HWSZ);
    const int ch = (int)((i / HWSZ) % HID);
    const int n  = (int)(i / ((size_t)HID * HWSZ));
    const float z = sigf(czr[((size_t)n * 256 + ch) * HWSZ + p]);
    const float q = tanhf(cq[i]);
    hout[i] = (1.f - z) * hin[i] + z * q;
}

// ============================================================================
extern "C" void kernel_launch(void* const* d_in, const int* in_sizes, int n_in,
                              void* d_out, int out_size, void* d_ws, size_t ws_size,
                              hipStream_t stream)
{
    (void)in_sizes; (void)n_in; (void)out_size; (void)ws_size;
    const float* h0  = (const float*)d_in[0];
    const float* x1  = (const float*)d_in[1];
    const float* x2  = (const float*)d_in[2];
    const float* wz1 = (const float*)d_in[3];  const float* bz1 = (const float*)d_in[4];
    const float* wr1 = (const float*)d_in[5];  const float* br1 = (const float*)d_in[6];
    const float* wq1 = (const float*)d_in[7];  const float* bq1 = (const float*)d_in[8];
    const float* wz2 = (const float*)d_in[9];  const float* bz2 = (const float*)d_in[10];
    const float* wr2 = (const float*)d_in[11]; const float* br2 = (const float*)d_in[12];
    const float* wq2 = (const float*)d_in[13]; const float* bq2 = (const float*)d_in[14];

    // ---- workspace carve-out ----
    char* base = (char*)d_ws;
    size_t off = 0;
    auto carve = [&](size_t bytes) -> void* {
        void* p = base + off;
        off += (bytes + 255) & ~(size_t)255;
        return p;
    };
    const size_t WPZR = (size_t)TAPS * CHUNKS * 256 * 32 * sizeof(bf16_t);
    const size_t WPQ  = (size_t)TAPS * CHUNKS * 128 * 32 * sizeof(bf16_t);
    const size_t CATB = (size_t)CINN * PIXT * sizeof(bf16_t);   // bf16 concat tensor
    const size_t CZRF = (size_t)256 * PIXT * sizeof(float);     // fused z|r conv out
    const size_t CQF  = (size_t)128 * PIXT * sizeof(float);
    const size_t HF   = (size_t)HID * PIXT * sizeof(float);

    bf16_t* wzr1p = (bf16_t*)carve(WPZR);
    bf16_t* wq1p  = (bf16_t*)carve(WPQ);
    bf16_t* wzr2p = (bf16_t*)carve(WPZR);
    bf16_t* wq2p  = (bf16_t*)carve(WPQ);
    float*  bzr1  = (float*) carve(256 * sizeof(float));
    float*  bzr2  = (float*) carve(256 * sizeof(float));
    bf16_t* mem_hx   = (bf16_t*)carve(CATB);
    bf16_t* mem_rhx  = (bf16_t*)carve(CATB);
    bf16_t* mem_hx2  = (bf16_t*)carve(CATB);
    bf16_t* mem_rhx2 = (bf16_t*)carve(CATB);
    float*  mem_czr1 = (float*) carve(CZRF);
    float*  mem_cq1  = (float*) carve(CQF);
    float*  mem_czr2 = (float*) carve(CZRF);
    float*  mem_cq2  = (float*) carve(CQF);
    bf16_t* din_bf   = (bf16_t*)carve(CATB);   // delta-step conv input (reused)
    float*  tmp_czr  = (float*) carve(CZRF);
    float*  tmp_cq   = (float*) carve(CQF);
    float*  hA = (float*)carve(HF);
    float*  hB = (float*)carve(HF);
    float*  hC = (float*)carve(HF);
    float*  h2 = (float*)d_out;

    const int PACK_B = (TAPS * CHUNKS * HID * 32 + 255) / 256;
    const int CAT_B  = (int)(((size_t)NB * CINN * HWSZ + 255) / 256);
    const int UPD_B  = (int)(((size_t)NB * HID * HWSZ + 255) / 256);
    const dim3 conv128(NB * HH * (WW / NT), 1);
    const dim3 conv256(NB * HH * (WW / NT), 2);

    // ---- weight / bias prep ----
    pack_weights<<<PACK_B, 256, 0, stream>>>(wz1, wzr1p, 0,   256);
    pack_weights<<<PACK_B, 256, 0, stream>>>(wr1, wzr1p, 128, 256);
    pack_weights<<<PACK_B, 256, 0, stream>>>(wq1, wq1p,  0,   128);
    pack_weights<<<PACK_B, 256, 0, stream>>>(wz2, wzr2p, 0,   256);
    pack_weights<<<PACK_B, 256, 0, stream>>>(wr2, wzr2p, 128, 256);
    pack_weights<<<PACK_B, 256, 0, stream>>>(wq2, wq2p,  0,   128);
    stack_bias<<<1, 256, 0, stream>>>(bz1, br1, bzr1);
    stack_bias<<<1, 256, 0, stream>>>(bz2, br2, bzr2);

    // ================= init step (x1) =================
    concat_init<<<CAT_B, 256, 0, stream>>>(h0, x1, mem_hx);
    conv_wmma<false><<<conv256, 256, 0, stream>>>(mem_hx, wzr1p, bzr1, nullptr, mem_czr1, 256);
    rhx_init<<<CAT_B, 256, 0, stream>>>(mem_czr1, h0, x1, mem_rhx);
    conv_wmma<false><<<conv128, 256, 0, stream>>>(mem_rhx, wq1p, bq1, nullptr, mem_cq1, 128);
    gru_update<<<UPD_B, 256, 0, stream>>>(mem_czr1, mem_cq1, h0, hA);

    concat_init<<<CAT_B, 256, 0, stream>>>(hA, x1, mem_hx2);
    conv_wmma<true><<<conv256, 256, 0, stream>>>(mem_hx2, wzr2p, bzr2, nullptr, mem_czr2, 256);
    rhx_init<<<CAT_B, 256, 0, stream>>>(mem_czr2, hA, x1, mem_rhx2);
    conv_wmma<true><<<conv128, 256, 0, stream>>>(mem_rhx2, wq2p, bq2, nullptr, mem_cq2, 128);
    gru_update<<<UPD_B, 256, 0, stream>>>(mem_czr2, mem_cq2, hA, hB);   // hB = h1

    // ================= delta step (x2) =================
    concat_delta<<<CAT_B, 256, 0, stream>>>(hB, x2, mem_hx, din_bf);
    conv_wmma<false><<<conv256, 256, 0, stream>>>(din_bf, wzr1p, nullptr, mem_czr1, tmp_czr, 256);
    rhx_delta<<<CAT_B, 256, 0, stream>>>(tmp_czr, hB, x2, mem_rhx, din_bf);
    conv_wmma<false><<<conv128, 256, 0, stream>>>(din_bf, wq1p, nullptr, mem_cq1, tmp_cq, 128);
    gru_update<<<UPD_B, 256, 0, stream>>>(tmp_czr, tmp_cq, hB, hC);

    concat_delta<<<CAT_B, 256, 0, stream>>>(hC, x2, mem_hx2, din_bf);
    conv_wmma<true><<<conv256, 256, 0, stream>>>(din_bf, wzr2p, nullptr, mem_czr2, tmp_czr, 256);
    rhx_delta<<<CAT_B, 256, 0, stream>>>(tmp_czr, hC, x2, mem_rhx2, din_bf);
    conv_wmma<true><<<conv128, 256, 0, stream>>>(din_bf, wq2p, nullptr, mem_cq2, tmp_cq, 128);
    gru_update<<<UPD_B, 256, 0, stream>>>(tmp_czr, tmp_cq, hC, h2);
}
